// PMC_56865366999524
// MI455X (gfx1250) — matrix operand order repaired and checked by reading, not compiled, not run on hardware
//
#include <hip/hip_runtime.h>

typedef __attribute__((ext_vector_type(16))) __bf16 v16bf;
typedef __attribute__((ext_vector_type(8)))  float  v8f;

#define BATCH 16384

union Frag {
    v16bf v;
    uint4 q[2];
};

__device__ __forceinline__ __bf16 f2bf(float f) {
    unsigned u = __float_as_uint(f);
    u += 0x7FFFu + ((u >> 16) & 1u);  // round-to-nearest-even
    unsigned short s = (unsigned short)(u >> 16);
    return __builtin_bit_cast(__bf16, s);
}

// ---------------------------------------------------------------------------
// CDNA5 async global->LDS copy (ASYNCcnt path), with sync fallback so the
// build never breaks if the builtin is absent on this toolchain.
// Builtin signature: (v4i addrspace(1)*, v4i addrspace(3)*, imm off, imm cpol)
// ---------------------------------------------------------------------------
#if defined(__AMDGCN__) && __has_builtin(__builtin_amdgcn_global_load_async_to_lds_b128)
#define HAVE_ASYNC_LDS 1
typedef int async_i4 __attribute__((vector_size(16)));
#endif

__device__ __forceinline__ void copy16_to_lds(const __bf16* g, __bf16* l) {
#ifdef HAVE_ASYNC_LDS
    __builtin_amdgcn_global_load_async_to_lds_b128(
        (__attribute__((address_space(1))) async_i4*)(g),
        (__attribute__((address_space(3))) async_i4*)(l), 0, 0);
#else
    *(uint4*)l = *(const uint4*)g;
#endif
}

__device__ __forceinline__ void wait_async_copies() {
#ifdef HAVE_ASYNC_LDS
#if __has_builtin(__builtin_amdgcn_s_wait_asynccnt)
    __builtin_amdgcn_s_wait_asynccnt(0);
#else
    asm volatile("s_wait_asynccnt 0" ::: "memory");
#endif
#endif
}

// ---------------------------------------------------------------------------
// Normalize obs -> x (bf16, padded to 256 cols) and x[:, :45] -> xs (64 cols)
// ---------------------------------------------------------------------------
__global__ __launch_bounds__(256)
void norm_kernel(const float* __restrict__ obs, const float* __restrict__ mean,
                 const float* __restrict__ var, __bf16* __restrict__ xbf,
                 __bf16* __restrict__ xsbf) {
    const int row = blockIdx.x;
    const int t = threadIdx.x;
    float v = 0.f;
    if (t < 235) v = (obs[(size_t)row * 235 + t] - mean[t]) * rsqrtf(var[t] + 1e-8f);
    xbf[(size_t)row * 256 + t] = f2bf(t < 235 ? v : 0.f);
    if (t < 64) xsbf[(size_t)row * 64 + t] = f2bf(t < 45 ? v : 0.f);
}

// ---------------------------------------------------------------------------
// Convert f32 weights to bf16 [N, Kpad] (optionally transposing), zero-padding
// K and N tails.  src element (n,k) = src[n*rs + k*cs].
// ---------------------------------------------------------------------------
__global__ __launch_bounds__(256)
void wconv_kernel(const float* __restrict__ src, __bf16* __restrict__ dst,
                  int N, int Nvalid, int K, int Kpad, int rs, int cs) {
    const int idx = blockIdx.x * 256 + threadIdx.x;
    if (idx >= N * Kpad) return;
    const int n = idx / Kpad;
    const int k = idx - n * Kpad;
    float v = (k < K && n < Nvalid) ? src[(size_t)n * rs + (size_t)k * cs] : 0.f;
    dst[idx] = f2bf(v);
}

// ---------------------------------------------------------------------------
// Codebook row squared norms (f32)
// ---------------------------------------------------------------------------
__global__ __launch_bounds__(256)
void cbnorm_kernel(const float* __restrict__ cb, float* __restrict__ out) {
    const int j = blockIdx.x * 256 + threadIdx.x;
    if (j >= 512) return;
    float s = 0.f;
    for (int k = 0; k < 64; ++k) {
        float v = cb[j * 64 + k];
        s += v * v;
    }
    out[j] = s;
}

// ---------------------------------------------------------------------------
// Register-blocked WMMA GEMM: C = act(A[B,K] @ Wt[N,K]^T + bias)
//   Block: 256 threads = 8 waves; covers 64 rows x 128 cols.
//   Wave: 4 row tiles x 1 col tile (B fragment reused 4x).
//   A staged in LDS in KC-column chunks via async global->LDS copies.
// ---------------------------------------------------------------------------
template <bool RELU, bool F32OUT>
__global__ __launch_bounds__(256)
void gemm_bf16(const __bf16* __restrict__ A, const __bf16* __restrict__ Wt,
               const float* __restrict__ bias, void* __restrict__ Cout,
               int K, int KC, int Ntiles, int Nvalid, int ldc, int coff) {
    extern __shared__ __bf16 sA[];  // 64 * KC elements
    const int rowBase = blockIdx.x * 64;
    const int wave = threadIdx.x >> 5;
    const int lane = threadIdx.x & 31;
    const int m = lane & 15;
    const int h = lane >> 4;
    const int colTile = blockIdx.y * 8 + wave;
    const bool active = colTile < Ntiles;  // wave-uniform

    v8f acc[4] = {};
    const __bf16* wrow = Wt + (size_t)(colTile * 16 + m) * K + h * 8;

    for (int kc = 0; kc < K; kc += KC) {
        if (kc) __syncthreads();  // previous chunk fully consumed
        // stage A[rowBase..rowBase+63, kc..kc+KC) -> LDS (async path)
        {
            const int c4 = KC >> 3;        // uint4s per row
            const int n4 = 64 * c4;
            for (int i = threadIdx.x; i < n4; i += 256) {
                const int r = i / c4;
                const int c = i - r * c4;
                copy16_to_lds(A + (size_t)(rowBase + r) * K + kc + c * 8,
                              sA + (size_t)r * KC + c * 8);
            }
            wait_async_copies();
        }
        __syncthreads();

        if (active) {
            for (int kb = 0; kb < KC; kb += 32) {
                Frag b;
                b.q[0] = *(const uint4*)(wrow + kc + kb);      // K = 8h + 0..7
                b.q[1] = *(const uint4*)(wrow + kc + kb + 16);  // K = 8h + 16..23
#pragma unroll
                for (int t = 0; t < 4; ++t) {
                    Frag a;
                    const __bf16* ar = sA + (size_t)(t * 16 + m) * KC + kb + h * 8;
                    a.q[0] = *(const uint4*)(ar);       // K = 8h + 0..7
                    a.q[1] = *(const uint4*)(ar + 16);  // K = 8h + 16..23
                    acc[t] = __builtin_amdgcn_wmma_f32_16x16x32_bf16(
                        false, a.v, false, b.v, (short)0, acc[t], false, false);
                }
            }
        }
    }

    if (active) {
        const int col = colTile * 16 + m;
        if (col < Nvalid) {
            const float bv = bias[col];
#pragma unroll
            for (int t = 0; t < 4; ++t) {
#pragma unroll
                for (int r = 0; r < 8; ++r) {
                    float v = acc[t][r] + bv;
                    if (RELU) v = fmaxf(v, 0.f);
                    const int row = rowBase + t * 16 + r + 8 * h;
                    if (F32OUT)
                        ((float*)Cout)[(size_t)row * ldc + coff + col] = v;
                    else
                        ((__bf16*)Cout)[(size_t)row * ldc + coff + col] = f2bf(v);
                }
            }
        }
    }
}

// ---------------------------------------------------------------------------
// VQ nearest-code: per wave, 16 rows of z_e; WMMA scores z_e @ cb^T over all
// 32 column tiles, fused argmin of (|c|^2 - 2 s) via wave32 shfl butterflies.
// ---------------------------------------------------------------------------
__global__ __launch_bounds__(256)
void vq_kernel(const __bf16* __restrict__ ze, const __bf16* __restrict__ cb,
               const float* __restrict__ cbn, int* __restrict__ nearest) {
    const int wave = threadIdx.x >> 5;
    const int lane = threadIdx.x & 31;
    const int rowTile = blockIdx.x * 8 + wave;
    const int m = lane & 15;
    const int h = lane >> 4;

    Frag a0, a1;
    const __bf16* arow = ze + (size_t)(rowTile * 16 + m) * 64 + h * 8;
    a0.q[0] = *(const uint4*)(arow);
    a0.q[1] = *(const uint4*)(arow + 16);
    a1.q[0] = *(const uint4*)(arow + 32);
    a1.q[1] = *(const uint4*)(arow + 48);

    float best[8];
    int bidx[8];
#pragma unroll
    for (int r = 0; r < 8; ++r) { best[r] = 3.4e38f; bidx[r] = 0; }

    for (int j = 0; j < 32; ++j) {
        Frag b0, b1;
        const __bf16* wrow = cb + (size_t)(j * 16 + m) * 64 + h * 8;
        b0.q[0] = *(const uint4*)(wrow);
        b0.q[1] = *(const uint4*)(wrow + 16);
        b1.q[0] = *(const uint4*)(wrow + 32);
        b1.q[1] = *(const uint4*)(wrow + 48);
        v8f acc = {};
        acc = __builtin_amdgcn_wmma_f32_16x16x32_bf16(false, a0.v, false, b0.v,
                                                      (short)0, acc, false, false);
        acc = __builtin_amdgcn_wmma_f32_16x16x32_bf16(false, a1.v, false, b1.v,
                                                      (short)0, acc, false, false);
        const int col = j * 16 + m;
        const float cn = cbn[col];
#pragma unroll
        for (int r = 0; r < 8; ++r) {
            const float d2 = cn - 2.f * acc[r];
            if (d2 < best[r]) { best[r] = d2; bidx[r] = col; }
        }
    }
#pragma unroll
    for (int mask = 1; mask < 16; mask <<= 1) {
#pragma unroll
        for (int r = 0; r < 8; ++r) {
            const float od = __shfl_xor(best[r], mask, 32);
            const int oi = __shfl_xor(bidx[r], mask, 32);
            if (od < best[r] || (od == best[r] && oi < bidx[r])) {
                best[r] = od;
                bidx[r] = oi;
            }
        }
    }
    if (m == 0) {
#pragma unroll
        for (int r = 0; r < 8; ++r) nearest[rowTile * 16 + r + 8 * h] = bidx[r];
    }
}

// ---------------------------------------------------------------------------
// Gather z_q = codebook[nearest] (forward value of st == z_q exactly)
// ---------------------------------------------------------------------------
__global__ __launch_bounds__(256)
void gather_kernel(const int* __restrict__ nearest, const float* __restrict__ cb,
                   __bf16* __restrict__ zq) {
    const int idx = blockIdx.x * 256 + threadIdx.x;  // BATCH*64 total
    const int row = idx >> 6;
    const int k = idx & 63;
    zq[idx] = f2bf(cb[(size_t)nearest[row] * 64 + k]);
}

// ---------------------------------------------------------------------------
extern "C" void kernel_launch(void* const* d_in, const int* in_sizes, int n_in,
                              void* d_out, int out_size, void* d_ws, size_t ws_size,
                              hipStream_t stream) {
    const float* obs      = (const float*)d_in[0];
    const float* rms_mean = (const float*)d_in[1];
    const float* rms_var  = (const float*)d_in[2];
    const float* enc_w1 = (const float*)d_in[3];   const float* enc_b1 = (const float*)d_in[4];
    const float* enc_w2 = (const float*)d_in[5];   const float* enc_b2 = (const float*)d_in[6];
    const float* enc_w3 = (const float*)d_in[7];   const float* enc_b3 = (const float*)d_in[8];
    const float* enc_w4 = (const float*)d_in[9];   const float* enc_b4 = (const float*)d_in[10];
    const float* codebook = (const float*)d_in[11];
    const float* zembd_w = (const float*)d_in[12]; const float* zembd_b = (const float*)d_in[13];
    const float* obsembd_w = (const float*)d_in[14]; const float* obsembd_b = (const float*)d_in[15];
    const float* dec_w1 = (const float*)d_in[16];  const float* dec_b1 = (const float*)d_in[17];
    const float* dec_w2 = (const float*)d_in[18];  const float* dec_b2 = (const float*)d_in[19];
    const float* dec_w3 = (const float*)d_in[20];  const float* dec_b3 = (const float*)d_in[21];

    char* ws = (char*)d_ws;
    size_t off = 0;
    auto alloc = [&](size_t bytes) -> char* {
        char* p = ws + off;
        off += (bytes + 255) & ~(size_t)255;
        return p;
    };

    __bf16* wt1 = (__bf16*)alloc((size_t)1024 * 256 * 2);
    __bf16* wt2 = (__bf16*)alloc((size_t)512 * 1024 * 2);
    __bf16* wt3 = (__bf16*)alloc((size_t)256 * 512 * 2);
    __bf16* wt4 = (__bf16*)alloc((size_t)64 * 256 * 2);
    __bf16* cbb = (__bf16*)alloc((size_t)512 * 64 * 2);
    __bf16* wtz = (__bf16*)alloc((size_t)128 * 64 * 2);
    __bf16* wto = (__bf16*)alloc((size_t)128 * 64 * 2);
    __bf16* wd1 = (__bf16*)alloc((size_t)512 * 256 * 2);
    __bf16* wd2 = (__bf16*)alloc((size_t)256 * 512 * 2);
    __bf16* wd3 = (__bf16*)alloc((size_t)16 * 256 * 2);
    float*  cbn = (float*)alloc(512 * 4);
    int*    nearest = (int*)alloc((size_t)BATCH * 4);
    __bf16* xs  = (__bf16*)alloc((size_t)BATCH * 64 * 2);
    __bf16* bufA = (__bf16*)alloc((size_t)BATCH * 1024 * 2);
    __bf16* bufB = (__bf16*)alloc((size_t)BATCH * 1024 * 2);

    auto wcv = [&](const float* src, __bf16* dst, int N, int Nv, int K, int Kpad,
                   int rs, int cs) {
        wconv_kernel<<<(N * Kpad + 255) / 256, 256, 0, stream>>>(src, dst, N, Nv, K, Kpad, rs, cs);
    };
    wcv(enc_w1, wt1, 1024, 1024, 235, 256, 1, 1024);
    wcv(enc_w2, wt2, 512, 512, 1024, 1024, 1, 512);
    wcv(enc_w3, wt3, 256, 256, 512, 512, 1, 256);
    wcv(enc_w4, wt4, 64, 64, 256, 256, 1, 64);
    wcv(codebook, cbb, 512, 512, 64, 64, 64, 1);  // already [N, K]
    wcv(zembd_w, wtz, 128, 128, 64, 64, 1, 128);
    wcv(obsembd_w, wto, 128, 128, 45, 64, 1, 128);
    wcv(dec_w1, wd1, 512, 512, 256, 256, 1, 512);
    wcv(dec_w2, wd2, 256, 256, 512, 512, 1, 256);
    wcv(dec_w3, wd3, 16, 12, 256, 256, 1, 12);
    cbnorm_kernel<<<2, 256, 0, stream>>>(codebook, cbn);

    norm_kernel<<<BATCH, 256, 0, stream>>>(obs, rms_mean, rms_var, bufA, xs);

    auto gemm = [&](const __bf16* A, const __bf16* Wt, const float* bias, void* C,
                    int K, int N, int Nvalid, int ldc, int coff, bool relu, bool f32out) {
        const int KC = (K < 256) ? K : 256;
        dim3 grid(BATCH / 64, (N / 16 + 7) / 8);
        size_t smem = (size_t)64 * KC * 2;
        if (relu)
            gemm_bf16<true, false><<<grid, 256, smem, stream>>>(A, Wt, bias, C, K, KC, N / 16, Nvalid, ldc, coff);
        else if (f32out)
            gemm_bf16<false, true><<<grid, 256, smem, stream>>>(A, Wt, bias, C, K, KC, N / 16, Nvalid, ldc, coff);
        else
            gemm_bf16<false, false><<<grid, 256, smem, stream>>>(A, Wt, bias, C, K, KC, N / 16, Nvalid, ldc, coff);
    };

    // encoder: x(bufA,256) -> h1(bufB,1024) -> h2(bufA,512) -> h3(bufB,256) -> z_e(bufA,64)
    gemm(bufA, wt1, enc_b1, bufB, 256, 1024, 1024, 1024, 0, true, false);
    gemm(bufB, wt2, enc_b2, bufA, 1024, 512, 512, 512, 0, true, false);
    gemm(bufA, wt3, enc_b3, bufB, 512, 256, 256, 256, 0, true, false);
    gemm(bufB, wt4, enc_b4, bufA, 256, 64, 64, 64, 0, false, false);

    // VQ: nearest code per row, then z_q (bufB)
    vq_kernel<<<(BATCH / 16) / 8, 256, 0, stream>>>(bufA, cbb, cbn, nearest);
    gather_kernel<<<(BATCH * 64) / 256, 256, 0, stream>>>(nearest, codebook, bufB);

    // dec_in (bufA, 256 wide): [z_q @ zembd | xs @ obsembd]
    gemm(bufB, wtz, zembd_b, bufA, 64, 128, 128, 256, 0, false, false);
    gemm(xs, wto, obsembd_b, bufA, 64, 128, 128, 256, 128, false, false);

    // decoder: dec_in(bufA,256) -> d1(bufB,512) -> d2(bufA,256) -> out(f32,12)
    gemm(bufA, wd1, dec_b1, bufB, 256, 512, 512, 512, 0, true, false);
    gemm(bufB, wd2, dec_b2, bufA, 512, 256, 256, 256, 0, true, false);
    gemm(bufA, wd3, dec_b3, d_out, 256, 16, 12, 12, 0, false, true);
}
